// MSDeformAttn_20581483282585
// MI455X (gfx1250) — matrix operand order repaired and checked by reading, not compiled, not run on hardware
//
#include <hip/hip_runtime.h>
#include <hip/hip_bf16.h>
#include <math.h>

// ---------------- problem constants (baked from reference) ----------------
#define D_MODEL 256
#define HEADS   8
#define LEVELS  4
#define POINTS  4
#define DPH     32              // D_MODEL / HEADS
#define S_TOT   19947
#define BATCH   2
#define M_REAL  (BATCH * S_TOT)         // 39894
#define M_PAD   39904                   // multiple of 32 (and 16)
#define KDIM    256
#define GEMM_GRID_Y 312                 // persistent M-loop stride (blocks in M)

typedef __attribute__((ext_vector_type(16))) _Float16 v16h;
typedef __attribute__((ext_vector_type(8)))  _Float16 v8h;
typedef __attribute__((ext_vector_type(8)))  float    v8f;
typedef __attribute__((ext_vector_type(4)))  int      v4i;

// pointer types for the async global->LDS builtin (int4 in AS1 / AS3)
typedef __attribute__((address_space(1))) v4i* as1_v4i_p;
typedef __attribute__((address_space(3))) v4i* as3_v4i_p;

union HFrag { v16h v; v8h h[2]; };

// Detect CDNA5 async global->LDS builtins (direct-to-LDS copy, ASYNCcnt-tracked)
#if defined(__has_builtin)
#if __has_builtin(__builtin_amdgcn_global_load_async_to_lds_b128) && \
    __has_builtin(__builtin_amdgcn_s_wait_asynccnt)
#define HAVE_ASYNC_LDS 1
#endif
#endif
#ifndef HAVE_ASYNC_LDS
#define HAVE_ASYNC_LDS 0
#endif

// ---------------- pack / convert kernels ----------------
__global__ void pack_a_kernel(const float* __restrict__ src, _Float16* __restrict__ dst,
                              int mreal, int mpad) {
    int idx = blockIdx.x * blockDim.x + threadIdx.x;
    int total = mpad * KDIM;
    if (idx >= total) return;
    int row = idx / KDIM;
    dst[idx] = (row < mreal) ? (_Float16)src[idx] : (_Float16)0.0f;
}

// W is (K, N) row-major fp32 -> Wt is (N, K) row-major f16
__global__ void pack_w_kernel(const float* __restrict__ W, _Float16* __restrict__ Wt,
                              int K, int N) {
    int idx = blockIdx.x * blockDim.x + threadIdx.x;
    if (idx >= K * N) return;
    int k = idx / N;
    int n = idx - k * N;
    Wt[(size_t)n * K + k] = (_Float16)W[idx];
}

// ---------------- WMMA GEMM: C(MxN) = A(MxK) * Wt(NxK)^T + bias ----------------
// Persistent blocks: fill 64KB weight panel into LDS once, hoist all B fragments
// into registers (weights are M-invariant), then loop M-tiles with only A loads
// + 16 WMMAs per 32x128 tile. 8 waves = 2(M) x 4(N); wave -> two 16x16 tiles.
__global__ __launch_bounds__(256)
void wmma_gemm_kernel(const _Float16* __restrict__ A, const _Float16* __restrict__ Bt,
                      const float* __restrict__ bias, float* __restrict__ C,
                      int N, int M_store) {
    __shared__ _Float16 Bs[128 * KDIM];   // 64 KB weight panel

    const _Float16* gpanel = Bt + (size_t)blockIdx.x * 128 * KDIM;
#if HAVE_ASYNC_LDS
    {
        // direct global->LDS async copy, 16B granules, ASYNCcnt-tracked
        for (int i = threadIdx.x; i < (128 * KDIM / 8); i += 256) {
            _Float16* gsrc = const_cast<_Float16*>(gpanel + (size_t)i * 8);
            _Float16* ldst = Bs + (size_t)i * 8;
            __builtin_amdgcn_global_load_async_to_lds_b128(
                (as1_v4i_p)gsrc, (as3_v4i_p)ldst, 0, 0);
        }
        __builtin_amdgcn_s_wait_asynccnt(0);
    }
#else
    {
        const float4* src = (const float4*)gpanel;
        float4* dst = (float4*)Bs;
        for (int i = threadIdx.x; i < (128 * KDIM * 2 / 16); i += 256)
            dst[i] = src[i];
    }
#endif
    __syncthreads();

    const int wave  = threadIdx.x >> 5;
    const int lane  = threadIdx.x & 31;
    const int wm    = wave >> 2;          // 0..1
    const int wn    = wave & 3;           // 0..3
    const int laneq = lane & 15;
    const int laneh = lane >> 4;

    // A fragment addressing (ISA 7.12.2, 16-bit A 16x32):
    //   lane holds row m0+laneq; halves [0..7] = K (k0 + laneh*8 .. +7),
    //   halves [8..15] = K (k0 + 16 + laneh*8 .. +7)
    const int ka = laneh * 8;

    // B fragment addressing (ISA 7.12.2 / sparse-B pattern, 16-bit B 32x16):
    //   lane holds col laneq; halves [0..7] = K (k0 + laneh*8 .. +7),
    //   halves [8..15] = K (k0 + 16 + laneh*8 .. +7)
    const int kb = laneh * 8;
    const _Float16* B0 = Bs + (size_t)(wn * 32 + laneq) * KDIM + kb;
    const _Float16* B1 = B0 + (size_t)16 * KDIM;

    // Hoist all B fragments into registers: 2 tiles x 8 k-steps x v16h = 128 VGPRs
    HFrag b0[8], b1[8];
    #pragma unroll
    for (int kk = 0; kk < 8; ++kk) {
        b0[kk].h[0] = *(const v8h*)(B0 + kk * 32);
        b0[kk].h[1] = *(const v8h*)(B0 + kk * 32 + 16);
        b1[kk].h[0] = *(const v8h*)(B1 + kk * 32);
        b1[kk].h[1] = *(const v8h*)(B1 + kk * 32 + 16);
    }

    const int n0   = blockIdx.x * 128 + wn * 32;
    const int col0 = n0 + laneq;
    const int col1 = col0 + 16;
    const float bb0 = bias[col0];
    const float bb1 = bias[col1];

    // Persistent loop over M tiles (stride = gridDim.y 32-row tiles)
    for (int mt = blockIdx.y; mt * 32 < M_PAD; mt += gridDim.y) {
        const int m0 = mt * 32 + wm * 16;
        const _Float16* Arow = A + (size_t)(m0 + laneq) * KDIM + ka;

        v8f c0 = {}; v8f c1 = {};
        #pragma unroll
        for (int kk = 0; kk < 8; ++kk) {
            HFrag a;
            a.h[0] = *(const v8h*)(Arow + kk * 32);
            a.h[1] = *(const v8h*)(Arow + kk * 32 + 16);
            c0 = __builtin_amdgcn_wmma_f32_16x16x32_f16(false, a.v, false, b0[kk].v,
                                                        (short)0, c0, false, false);
            c1 = __builtin_amdgcn_wmma_f32_16x16x32_f16(false, a.v, false, b1[kk].v,
                                                        (short)0, c1, false, false);
        }

        // C/D layout: VGPR i -> (M = m0 + i + 8*laneh, N = col)
        const int rowb = m0 + laneh * 8;
        #pragma unroll
        for (int i = 0; i < 8; ++i) {
            int row = rowb + i;
            if (row < M_store) {
                C[(size_t)row * N + col0] = c0[i] + bb0;
                C[(size_t)row * N + col1] = c1[i] + bb1;
            }
        }
    }
}

// ---------------- softmax over 16 (level,point) logits per head ----------------
__global__ void softmax16_kernel(float* __restrict__ aw, int mreal) {
    int t = blockIdx.x * blockDim.x + threadIdx.x;
    int row = t >> 3;
    int h   = t & 7;
    if (row >= mreal) return;
    float* p = aw + (size_t)row * (HEADS * LEVELS * POINTS) + h * (LEVELS * POINTS);
    float mx = p[0];
    #pragma unroll
    for (int i = 1; i < 16; ++i) mx = fmaxf(mx, p[i]);
    float e[16], s = 0.0f;
    #pragma unroll
    for (int i = 0; i < 16; ++i) { e[i] = expf(p[i] - mx); s += e[i]; }
    float inv = 1.0f / s;
    #pragma unroll
    for (int i = 0; i < 16; ++i) p[i] = e[i] * inv;
}

// ---------------- deformable sampling: one wave per (query, head), lane = channel ----------------
__global__ __launch_bounds__(256)
void sample_kernel(const float* __restrict__ vbuf,     // (M_PAD, 256)  value projection
                   const float* __restrict__ offs,     // (M_PAD, 256)  sampling offsets
                   const float* __restrict__ aw,       // (M_PAD, 128)  softmaxed attn weights
                   const float* __restrict__ refp,     // (B, S, LEVELS, 2)
                   _Float16* __restrict__ attn) {      // (M_PAD, 256)  f16 output for final GEMM
    const int q = blockIdx.x;          // 0 .. M_PAD-1
    const int h = threadIdx.x >> 5;    // head
    const int d = threadIdx.x & 31;    // channel within head

    const int Hs[LEVELS]     = {100, 50, 25, 13};
    const int Ws[LEVELS]     = {150, 75, 38, 19};
    const int starts[LEVELS] = {0, 15000, 18750, 19700};

    float acc = 0.0f;
    if (q < M_REAL) {
        const int b = q / S_TOT;
        const float* offr = offs + (size_t)q * D_MODEL + h * (LEVELS * POINTS * 2);
        const float* awr  = aw   + (size_t)q * (HEADS * LEVELS * POINTS) + h * (LEVELS * POINTS);
        const float* ref  = refp + (size_t)q * (LEVELS * 2);
        const float* vch  = vbuf + h * DPH + d;   // channel-offset base

        #pragma unroll
        for (int lvl = 0; lvl < LEVELS; ++lvl) {
            const int Hh = Hs[lvl], Ww = Ws[lvl];
            const float fH = (float)Hh, fW = (float)Ww;
            const size_t rowbase = (size_t)(b * S_TOT + starts[lvl]);
            const float rx = ref[lvl * 2 + 0];
            const float ry = ref[lvl * 2 + 1];
            #pragma unroll
            for (int p = 0; p < POINTS; ++p) {
                const float ox = offr[(lvl * POINTS + p) * 2 + 0];
                const float oy = offr[(lvl * POINTS + p) * 2 + 1];
                const float wat = awr[lvl * POINTS + p];
                // (ref + off/size) * size - 0.5
                const float px = rx * fW + ox - 0.5f;
                const float py = ry * fH + oy - 0.5f;
                const float fx = floorf(px), fy = floorf(py);
                const float wx = px - fx,    wy = py - fy;
                const int x0 = (int)fx, y0 = (int)fy;
                const int x1 = x0 + 1,  y1 = y0 + 1;

                float g00 = 0.f, g01 = 0.f, g10 = 0.f, g11 = 0.f;
                const bool vx0 = (x0 >= 0) & (x0 < Ww);
                const bool vx1 = (x1 >= 0) & (x1 < Ww);
                const bool vy0 = (y0 >= 0) & (y0 < Hh);
                const bool vy1 = (y1 >= 0) & (y1 < Hh);
                if (vy0 & vx0) g00 = vch[(rowbase + (size_t)(y0 * Ww + x0)) * D_MODEL];
                if (vy0 & vx1) g01 = vch[(rowbase + (size_t)(y0 * Ww + x1)) * D_MODEL];
                if (vy1 & vx0) g10 = vch[(rowbase + (size_t)(y1 * Ww + x0)) * D_MODEL];
                if (vy1 & vx1) g11 = vch[(rowbase + (size_t)(y1 * Ww + x1)) * D_MODEL];

                const float top = g00 * (1.0f - wx) + g01 * wx;
                const float bot = g10 * (1.0f - wx) + g11 * wx;
                acc += wat * (top * (1.0f - wy) + bot * wy);
            }
        }
    }
    attn[(size_t)q * D_MODEL + h * DPH + d] = (_Float16)acc;
}

// ---------------- launch ----------------
extern "C" void kernel_launch(void* const* d_in, const int* in_sizes, int n_in,
                              void* d_out, int out_size, void* d_ws, size_t ws_size,
                              hipStream_t stream) {
    const float* query = (const float*)d_in[0];
    const float* refp  = (const float*)d_in[1];
    const float* value = (const float*)d_in[2];
    // d_in[3] = spatial_shapes (static, baked as constants)
    const float* Wv  = (const float*)d_in[4];
    const float* bv  = (const float*)d_in[5];
    const float* Wso = (const float*)d_in[6];
    const float* bso = (const float*)d_in[7];
    const float* Waw = (const float*)d_in[8];
    const float* baw = (const float*)d_in[9];
    const float* Wo  = (const float*)d_in[10];
    const float* bo  = (const float*)d_in[11];

    char* ws = (char*)d_ws;
    // workspace carving (all sizes multiple of 256 B)
    const size_t szAh   = (size_t)M_PAD * D_MODEL * sizeof(_Float16);  // 20.4 MB
    const size_t szF32  = (size_t)M_PAD * D_MODEL * sizeof(float);     // 40.9 MB
    const size_t szAw   = (size_t)M_PAD * (HEADS * LEVELS * POINTS) * sizeof(float); // 20.4 MB
    _Float16* qh    = (_Float16*)(ws);                       ws += szAh;
    _Float16* vh    = (_Float16*)(ws);                       ws += szAh; // reused as attn f16
    float*    vbuf  = (float*)(ws);                          ws += szF32;
    float*    offsb = (float*)(ws);                          ws += szF32;
    float*    awb   = (float*)(ws);                          ws += szAw;
    _Float16* WvT   = (_Float16*)(ws);                       ws += (size_t)256 * 256 * 2;
    _Float16* WsoT  = (_Float16*)(ws);                       ws += (size_t)256 * 256 * 2;
    _Float16* WawT  = (_Float16*)(ws);                       ws += (size_t)256 * 128 * 2;
    _Float16* WoT   = (_Float16*)(ws);
    _Float16* attn  = vh;   // value-f16 panel is dead after GEMM1; reuse for attention output

    const int packBlocks = (M_PAD * D_MODEL + 255) / 256;
    pack_a_kernel<<<packBlocks, 256, 0, stream>>>(query, qh, M_REAL, M_PAD);
    pack_a_kernel<<<packBlocks, 256, 0, stream>>>(value, vh, M_REAL, M_PAD);
    pack_w_kernel<<<(256 * 256 + 255) / 256, 256, 0, stream>>>(Wv,  WvT,  256, 256);
    pack_w_kernel<<<(256 * 256 + 255) / 256, 256, 0, stream>>>(Wso, WsoT, 256, 256);
    pack_w_kernel<<<(256 * 128 + 255) / 256, 256, 0, stream>>>(Waw, WawT, 256, 128);
    pack_w_kernel<<<(256 * 256 + 255) / 256, 256, 0, stream>>>(Wo,  WoT,  256, 256);

    const dim3 g256(2, GEMM_GRID_Y), g128(1, GEMM_GRID_Y);
    // v = value @ Wv + bv   (store padded rows too; sampling never reads them)
    wmma_gemm_kernel<<<g256, 256, 0, stream>>>(vh, WvT, bv, vbuf, 256, M_PAD);
    // offs = query @ Wso + bso
    wmma_gemm_kernel<<<g256, 256, 0, stream>>>(qh, WsoT, bso, offsb, 256, M_PAD);
    // aw logits = query @ Waw + baw
    wmma_gemm_kernel<<<g128, 256, 0, stream>>>(qh, WawT, baw, awb, 128, M_PAD);
    // softmax over each head's 16 logits
    softmax16_kernel<<<(M_REAL * HEADS + 255) / 256, 256, 0, stream>>>(awb, M_REAL);
    // deformable bilinear sampling -> f16 attention output (padded rows zeroed)
    sample_kernel<<<M_PAD, 256, 0, stream>>>(vbuf, offsb, awb, refp, attn);
    // out = attn @ Wo + bo  -> d_out (exactly M_REAL x 256)
    wmma_gemm_kernel<<<g256, 256, 0, stream>>>(attn, WoT, bo, (float*)d_out, 256, M_REAL);

    (void)in_sizes; (void)n_in; (void)out_size; (void)ws_size;
}
